// VqaPrototypeModel_26268019982523
// MI455X (gfx1250) — matrix operand order repaired and checked by reading, not compiled, not run on hardware
//
#include <hip/hip_runtime.h>
#include <hip/hip_bf16.h>

// ---------------------------------------------------------------------------
// Problem constants (B=16, S=38, D=1792, H=4, DK=448, NP=128, INF=768)
// ---------------------------------------------------------------------------
#define ROWS   608      // B*S
#define DMODEL 1792
#define NPROTO 128
#define NHEAD  4
#define DKHEAD 448
#define INFEAT 768

typedef __attribute__((ext_vector_type(16))) _Float16 v16h;
typedef __attribute__((ext_vector_type(8)))  _Float16 v8h;
typedef __attribute__((ext_vector_type(8)))  float    v8f;
typedef int v4i_vs __attribute__((vector_size(16)));   // matches builtin's pointee

// ---------------------------------------------------------------------------
// CDNA5 async global->LDS copy (ASYNCcnt-tracked), with sync fallback.
// ---------------------------------------------------------------------------
#if defined(__has_builtin)
# if __has_builtin(__builtin_amdgcn_global_load_async_to_lds_b128)
#  define USE_ASYNC_LDS 1
# endif
# if __has_builtin(__builtin_amdgcn_s_wait_asynccnt)
#  define WAIT_ASYNC(n) __builtin_amdgcn_s_wait_asynccnt(n)
# endif
#endif
#ifndef WAIT_ASYNC
# define WAIT_ASYNC(n) asm volatile("s_wait_asynccnt %0" :: "i"(n) : "memory")
#endif

#if USE_ASYNC_LDS
__device__ __forceinline__ void async_copy16(const float* g, float* l)
{
    __builtin_amdgcn_global_load_async_to_lds_b128(
        (__attribute__((address_space(1))) v4i_vs*)(g),
        (__attribute__((address_space(3))) v4i_vs*)(l),
        0, 0);
}
#endif

// ---------------------------------------------------------------------------
// Generic GEMM: C[M,N] = concat_K(A1[:, :K1], A2[:, :K-K1]) @ W[K,N] + bias[N]
// f32 in/out, f16 WMMA core (v_wmma_f32_16x16x32_f16), f32 accumulate.
// Block: 256 threads (8 waves), tile 64(M) x 128(N), K-step 32.
// Wave w: wm = w&3 (16-row slab), wn = w>>2 (64-col slab -> 4 accumulators).
// Double-buffered LDS, tiles staged raw-f32 by async global->LDS DMA.
// ---------------------------------------------------------------------------
__global__ void __launch_bounds__(256)
gemm_wmma_f16(const float* __restrict__ A1, const float* __restrict__ A2,
              const float* __restrict__ W,  const float* __restrict__ bias,
              float* __restrict__ C,
              int M, int N, int K, int K1, int lda1, int lda2)
{
    __shared__ float LdsA[2][64 * 32];    // [row][k], f32
    __shared__ float LdsB[2][32 * 128];   // [k][n],  f32

    const int tid  = threadIdx.x;
    const int lane = tid & 31;
    const int w    = tid >> 5;
    const int wm   = w & 3;
    const int wn   = w >> 2;
    const int m0   = blockIdx.x * 64;
    const int n0   = blockIdx.y * 128;

    // Stage one 64x32 A tile: 512 float4 chunks, 2 per thread.
    auto stageA = [&](int k0, float* dst) {
        #pragma unroll
        for (int j = 0; j < 2; ++j) {
            const int c    = tid + j * 256;
            const int r    = c >> 3;
            const int koff = (c & 7) << 2;
            int gm = m0 + r; if (gm >= M) gm = M - 1;   // clamp: rows>=M unused
            const int gk = k0 + koff;
            const float* src = (gk < K1) ? (A1 + (size_t)gm * lda1 + gk)
                                         : (A2 + (size_t)gm * lda2 + (gk - K1));
#if USE_ASYNC_LDS
            async_copy16(src, dst + c * 4);
#else
            *(float4*)(dst + c * 4) = *(const float4*)src;
#endif
        }
    };
    // Stage one 32x128 B tile: 1024 float4 chunks, 4 per thread.
    auto stageB = [&](int k0, float* dst) {
        #pragma unroll
        for (int j = 0; j < 4; ++j) {
            const int c    = tid + j * 256;
            const int kr   = c >> 5;
            const int noff = (c & 31) << 2;
            const float* src = W + (size_t)(k0 + kr) * N + n0 + noff;
#if USE_ASYNC_LDS
            async_copy16(src, dst + c * 4);
#else
            *(float4*)(dst + c * 4) = *(const float4*)src;
#endif
        }
    };

    // Fragment addressing.
    // A (contiguous k order, same as B): lane L -> row L%16; halves 0..15 = K klo..klo+15;
    //   klo = 16 for lanes 16..31.
    const int arow = wm * 16 + (lane & 15);
    const int aklo = (lane >= 16) ? 16 : 0;
    // B (contiguous k order, same as A): lane L -> col L%16; halves 0..15 = K bk0..bk0+15.
    const int bn   = wn * 64 + (lane & 15);
    const int bk0  = (lane >= 16) ? 16 : 0;

    v8f acc[4] = {v8f{}, v8f{}, v8f{}, v8f{}};

    const int nsteps = K / 32;
    stageA(0, LdsA[0]);
    stageB(0, LdsB[0]);

    for (int s = 0; s < nsteps; ++s) {
        const int cur = s & 1;
        if (s + 1 < nsteps) {
            stageA((s + 1) * 32, LdsA[cur ^ 1]);
            stageB((s + 1) * 32, LdsB[cur ^ 1]);
#if USE_ASYNC_LDS
            WAIT_ASYNC(6);   // allow the 6 just-issued copies to stay in flight
#endif
        } else {
#if USE_ASYNC_LDS
            WAIT_ASYNC(0);
#endif
        }
        __syncthreads();

        const float* Ab = LdsA[cur];
        const float* Bb = LdsB[cur];

        // A fragment: 4x ds_load_b128 + f32->f16 convert
        const float4* ap  = (const float4*)(Ab + arow * 32 + aklo);
        const float4* ap2 = (const float4*)(Ab + arow * 32 + aklo + 8);
        const float4 a0 = ap[0],  a1 = ap[1];
        const float4 a2 = ap2[0], a3 = ap2[1];
        v16h a;
        a[0] = (_Float16)a0.x; a[1] = (_Float16)a0.y; a[2] = (_Float16)a0.z; a[3] = (_Float16)a0.w;
        a[4] = (_Float16)a1.x; a[5] = (_Float16)a1.y; a[6] = (_Float16)a1.z; a[7] = (_Float16)a1.w;
        a[8] = (_Float16)a2.x; a[9] = (_Float16)a2.y; a[10] = (_Float16)a2.z; a[11] = (_Float16)a2.w;
        a[12] = (_Float16)a3.x; a[13] = (_Float16)a3.y; a[14] = (_Float16)a3.z; a[15] = (_Float16)a3.w;

        #pragma unroll
        for (int j = 0; j < 4; ++j) {
            const int nl = bn + j * 16;
            v16h b;
            #pragma unroll
            for (int h = 0; h < 16; ++h)
                b[h] = (_Float16)Bb[(bk0 + h) * 128 + nl];
            acc[j] = __builtin_amdgcn_wmma_f32_16x16x32_f16(
                         false, a, false, b, (short)0, acc[j], false, false);
        }
        __syncthreads();
    }

    // Store (16x16 f32 C/D layout: vgpr g -> row g, +8 for lanes 16..31).
    const int crow0 = m0 + wm * 16 + ((lane >= 16) ? 8 : 0);
    const int ncol0 = n0 + wn * 64 + (lane & 15);
    float bia[4];
    #pragma unroll
    for (int j = 0; j < 4; ++j) bia[j] = bias[ncol0 + j * 16];
    #pragma unroll
    for (int g = 0; g < 8; ++g) {
        const int crow = crow0 + g;
        if (crow < M) {
            #pragma unroll
            for (int j = 0; j < 4; ++j)
                C[(size_t)crow * N + ncol0 + j * 16] = acc[j][g] + bia[j];
        }
    }
}

// ---------------------------------------------------------------------------
// Attention collapse: per (row, head) find argmax_p <q_h, k_h[p]> over 128
// prototypes, then X[row, head] = Vp[argmax, head].  (top-3 of the tiled
// score vector are three copies of this max; softmax weights = 1/3 each.)
// One block per row (128 threads = 4 waves = 4 heads), 4 prototypes per lane.
// ---------------------------------------------------------------------------
__global__ void __launch_bounds__(128)
attn_argmax_gather(const float* __restrict__ Q, const float* __restrict__ Kp,
                   const float* __restrict__ Vp, float* __restrict__ X)
{
    __shared__ __align__(16) float qs[DMODEL];
    const int row  = blockIdx.x;
    const int tid  = threadIdx.x;
    const int lane = tid & 31;
    const int h    = tid >> 5;
    const int hb   = h * DKHEAD;

    for (int i = tid; i < DMODEL; i += 128)
        qs[i] = Q[(size_t)row * DMODEL + i];
    __syncthreads();

    float best = -3.4e38f;
    int   bi   = 0;
    const float4* q4 = (const float4*)(qs + hb);
    for (int j = 0; j < 4; ++j) {
        const int p = j * 32 + lane;
        const float4* kp4 = (const float4*)(Kp + (size_t)p * DMODEL + hb);
        float s = 0.0f;
        for (int t = 0; t < DKHEAD / 4; ++t) {
            const float4 kv = kp4[t];
            const float4 qv = q4[t];
            s += kv.x * qv.x + kv.y * qv.y + kv.z * qv.z + kv.w * qv.w;
        }
        if (s > best) { best = s; bi = p; }   // ascending p => ties keep low idx
    }
    // wave32 max-argmax reduction, lowest index wins ties (top_k order)
    for (int off = 16; off > 0; off >>= 1) {
        const float ob = __shfl_xor(best, off, 32);
        const int   oi = __shfl_xor(bi,   off, 32);
        if (ob > best || (ob == best && oi < bi)) { best = ob; bi = oi; }
    }

    const float* vrow = Vp + (size_t)bi * DMODEL + hb;
    float*       xrow = X  + (size_t)row * DMODEL + hb;
    for (int i = lane; i < DKHEAD; i += 32)
        xrow[i] = vrow[i];
}

// ---------------------------------------------------------------------------
// logits[row, 0:2] = reduced[row, :] @ Wqa[768, 2] + bqa.
// 8 waves per block, one row per wave, shuffle-reduce.
// ---------------------------------------------------------------------------
__global__ void __launch_bounds__(256)
qa_logits(const float* __restrict__ Red, const float* __restrict__ Wqa,
          const float* __restrict__ bqa, float* __restrict__ out)
{
    const int lane = threadIdx.x & 31;
    const int w    = threadIdx.x >> 5;
    const int row  = blockIdx.x * 8 + w;
    const float* r = Red + (size_t)row * INFEAT;
    float s0 = 0.0f, s1 = 0.0f;
    for (int i = lane; i < INFEAT; i += 32) {
        const float rv = r[i];
        s0 += rv * Wqa[i * 2 + 0];
        s1 += rv * Wqa[i * 2 + 1];
    }
    for (int off = 16; off > 0; off >>= 1) {
        s0 += __shfl_xor(s0, off, 32);
        s1 += __shfl_xor(s1, off, 32);
    }
    if (lane == 0) {
        out[row * 2 + 0] = s0 + bqa[0];
        out[row * 2 + 1] = s1 + bqa[1];
    }
}

// ---------------------------------------------------------------------------
extern "C" void kernel_launch(void* const* d_in, const int* in_sizes, int n_in,
                              void* d_out, int out_size, void* d_ws, size_t ws_size,
                              hipStream_t stream)
{
    const float* cf    = (const float*)d_in[0];   // [608, 1792]
    const float* proto = (const float*)d_in[1];   // [128, 1792]
    // d_in[2] = attention_mask (unused by the reference)
    const float* Wq  = (const float*)d_in[3];
    const float* bq  = (const float*)d_in[4];
    const float* Wk  = (const float*)d_in[5];
    const float* bk  = (const float*)d_in[6];
    const float* Wv  = (const float*)d_in[7];
    const float* bv  = (const float*)d_in[8];
    const float* Wo  = (const float*)d_in[9];
    const float* bo  = (const float*)d_in[10];
    const float* Wfc = (const float*)d_in[11];    // [3584, 768]
    const float* bfc = (const float*)d_in[12];
    const float* Wqa = (const float*)d_in[13];    // [768, 2]
    const float* bqa = (const float*)d_in[14];
    float* out = (float*)d_out;                   // [608, 2]

    // workspace layout (16 MB of f32, fully overwritten every call)
    float* Q   = (float*)d_ws;                       // 608*1792
    float* Kp  = Q   + (size_t)ROWS * DMODEL;        // 128*1792
    float* Vp  = Kp  + (size_t)NPROTO * DMODEL;      // 128*1792
    float* X   = Vp  + (size_t)NPROTO * DMODEL;      // 608*1792
    float* R   = X   + (size_t)ROWS * DMODEL;        // 608*1792
    float* Red = R   + (size_t)ROWS * DMODEL;        // 608*768

    const dim3 blk(256);

    // Q = cf @ Wq + bq
    gemm_wmma_f16<<<dim3((ROWS + 63) / 64, DMODEL / 128), blk, 0, stream>>>(
        cf, cf, Wq, bq, Q, ROWS, DMODEL, DMODEL, DMODEL, DMODEL, DMODEL);
    // Kp = proto @ Wk + bk   (only the 128 distinct prototype rows)
    gemm_wmma_f16<<<dim3(NPROTO / 64, DMODEL / 128), blk, 0, stream>>>(
        proto, proto, Wk, bk, Kp, NPROTO, DMODEL, DMODEL, DMODEL, DMODEL, DMODEL);
    // Vp = proto @ Wv + bv
    gemm_wmma_f16<<<dim3(NPROTO / 64, DMODEL / 128), blk, 0, stream>>>(
        proto, proto, Wv, bv, Vp, NPROTO, DMODEL, DMODEL, DMODEL, DMODEL, DMODEL);
    // X = gather(Vp, argmax_p <Q_h, Kp_h>)
    attn_argmax_gather<<<ROWS, 128, 0, stream>>>(Q, Kp, Vp, X);
    // R = X @ Wo + bo
    gemm_wmma_f16<<<dim3((ROWS + 63) / 64, DMODEL / 128), blk, 0, stream>>>(
        X, X, Wo, bo, R, ROWS, DMODEL, DMODEL, DMODEL, DMODEL, DMODEL);
    // Red = [cf | R] @ Wfc + bfc  (K split at 1792, no concat copy)
    gemm_wmma_f16<<<dim3((ROWS + 63) / 64, INFEAT / 128), blk, 0, stream>>>(
        cf, R, Wfc, bfc, Red, ROWS, INFEAT, 2 * DMODEL, DMODEL, DMODEL, DMODEL);
    // logits = Red @ Wqa + bqa
    qa_logits<<<ROWS / 8, 256, 0, stream>>>(Red, Wqa, bqa, out);
}
